// Joiner_60593398612577
// MI455X (gfx1250) — matrix operand order repaired and checked by reading, not compiled, not run on hardware
//
#include <hip/hip_runtime.h>

typedef __attribute__((ext_vector_type(16))) __bf16 v16bf;
typedef __attribute__((ext_vector_type(8)))  float  v8f;

#define R_TOTAL 32000      // N*T*S rows
#define CDIM    512        // ENC = DEC = JOINT
#define KT      16         // 512 / 32 k-tiles
#define OV      1999       // out vocab
#define VFULL   2000       // final vocab (blank inserted at col 0)
#define NPADV   2048       // padded vocab N for full 16-col tiles (4 slabs x 512)

__device__ __forceinline__ unsigned short f2bf(float f) {
  union { float f; unsigned int u; } v; v.f = f;
  unsigned int u = v.u;
  u += 0x7fffu + ((u >> 16) & 1u);   // round to nearest even
  return (unsigned short)(u >> 16);
}

// A-matrix (16x32 bf16) wave32 swizzle, per ISA 7.12.2:
// lane = m + 16*lgrp ; halves 0..7 -> K = k0 + lgrp*8 + h ; halves 8..15 -> K = k0+16+lgrp*8+(h-8)
__device__ __forceinline__ size_t a_swz(int r, int k) {
  int m = r & 15, rt = r >> 4;
  int kt = k >> 5, kk = k & 31;
  int lgrp = (kk >> 3) & 1;
  int h = (kk & 7) + ((kk >> 4) << 3);
  int lane = m + (lgrp << 4);
  return ((size_t)(rt * KT + kt) << 9) + (lane << 4) + h;
}

// B-matrix (32x16 bf16) wave32 swizzle: lanes 0-15 hold K=0..7,16..23, lanes 16-31 hold K=8..15,24..31
__device__ __forceinline__ size_t b_swz(int n, int k) {
  int c = n & 15, nt = n >> 4;
  int kt = k >> 5, kk = k & 31;
  int lane = c + (((kk >> 3) & 1) << 4);
  int h = (kk & 7) + ((kk >> 4) << 3);
  return ((size_t)(nt * KT + kt) << 9) + (lane << 4) + h;
}

// ---- fp32 weights [nrows x 512] -> bf16 B-swizzled [npad x 512], zero-padded ----
__global__ __launch_bounds__(256) void swizzle_w(const float* __restrict__ W,
                                                 unsigned short* __restrict__ Wb,
                                                 int nrows, int npad) {
  int idx = blockIdx.x * 256 + threadIdx.x;
  if (idx >= npad * CDIM) return;
  int n = idx >> 9, k = idx & (CDIM - 1);
  float v = (n < nrows) ? W[(size_t)n * CDIM + k] : 0.0f;
  Wb[b_swz(n, k)] = f2bf(v);
}

// ---- fp32 activations [R x 512] -> bf16 A-swizzled; optional fused BiasNorm ----
__global__ __launch_bounds__(256) void swizzle_a(const float* __restrict__ X,
                                                 unsigned short* __restrict__ As,
                                                 const float* __restrict__ nb,
                                                 const float* __restrict__ nls,
                                                 int do_norm) {
  int lane = threadIdx.x & 31;
  int wave = threadIdx.x >> 5;          // one wave per row, 16 elems per lane
  int r = blockIdx.x * 8 + wave;
  int kbase = lane << 4;
  const float* xr = X + (size_t)r * CDIM + kbase;
  float x[16];
  #pragma unroll
  for (int i = 0; i < 16; ++i) x[i] = xr[i];
  float scale = 1.0f;
  if (do_norm) {
    float ss = 0.0f;
    #pragma unroll
    for (int i = 0; i < 16; ++i) { float d = x[i] - nb[kbase + i]; ss += d * d; }
    ss += __shfl_xor(ss, 1, 32);
    ss += __shfl_xor(ss, 2, 32);
    ss += __shfl_xor(ss, 4, 32);
    ss += __shfl_xor(ss, 8, 32);
    ss += __shfl_xor(ss, 16, 32);
    scale = __expf(nls[0]) * rsqrtf(ss * (1.0f / 512.0f));
  }
  #pragma unroll
  for (int i = 0; i < 16; ++i) As[a_swz(r, kbase + i)] = f2bf(x[i] * scale);
}

// ---- 512x512 GEMM + bias + BiasNorm; optional fused blank head ----
// WG = 256 thr = 8 waves; 16 rows x 512 cols; wave w owns cols [w*64, w*64+64)
__global__ __launch_bounds__(256) void gemm512_bn(
    const unsigned short* __restrict__ As,
    const unsigned short* __restrict__ Wb,
    const float* __restrict__ lin_b,
    const float* __restrict__ bn_b,
    const float* __restrict__ bn_ls,
    float* __restrict__ Yout,
    const float* __restrict__ Yother,      // enc (for blank head)
    const float* __restrict__ w_blank,
    const float* __restrict__ b_blank,
    float* __restrict__ out,               // [R, VFULL]
    int fuse_blank)
{
  int lane = threadIdx.x & 31;
  int wave = threadIdx.x >> 5;
  int rt = blockIdx.x;

  v8f acc[4] = {};
  const unsigned short* Ab = As + ((size_t)(rt * KT) << 9);
  #pragma unroll 2
  for (int kt = 0; kt < KT; ++kt) {
    v16bf a = *((const v16bf*)(Ab + ((size_t)kt << 9)) + lane);
    #pragma unroll
    for (int t = 0; t < 4; ++t) {
      int ntile = wave * 4 + t;
      v16bf b = *((const v16bf*)(Wb + ((size_t)(ntile * KT + kt) << 9)) + lane);
      acc[t] = __builtin_amdgcn_wmma_f32_16x16x32_bf16(
          false, a, false, b, (short)0, acc[t], false, false);
    }
  }

  int mbase = (lane >> 4) << 3;           // rows 0..7 or 8..15
  int nb0 = wave * 64 + (lane & 15);

  float y[4][8];
  #pragma unroll
  for (int t = 0; t < 4; ++t) {
    float lb = lin_b[nb0 + t * 16];
    #pragma unroll
    for (int v = 0; v < 8; ++v) y[t][v] = acc[t][v] + lb;
  }

  __shared__ float red[16][8];
  __shared__ float ssrow[16];

  float pp[8];
  #pragma unroll
  for (int v = 0; v < 8; ++v) {
    float p = 0.0f;
    #pragma unroll
    for (int t = 0; t < 4; ++t) {
      float d = y[t][v] - bn_b[nb0 + t * 16];
      p += d * d;
    }
    p += __shfl_xor(p, 1, 32);
    p += __shfl_xor(p, 2, 32);
    p += __shfl_xor(p, 4, 32);
    p += __shfl_xor(p, 8, 32);
    pp[v] = p;
  }
  if ((lane & 15) == 0) {
    #pragma unroll
    for (int v = 0; v < 8; ++v) red[mbase + v][wave] = pp[v];
  }
  __syncthreads();
  if (threadIdx.x < 16) {
    float s = 0.0f;
    #pragma unroll
    for (int w = 0; w < 8; ++w) s += red[threadIdx.x][w];
    ssrow[threadIdx.x] = s;
  }
  __syncthreads();

  float els = __expf(bn_ls[0]);
  float sc[8];
  #pragma unroll
  for (int v = 0; v < 8; ++v) sc[v] = els * rsqrtf(ssrow[mbase + v] * (1.0f / 512.0f));

  if (!fuse_blank) {
    #pragma unroll
    for (int t = 0; t < 4; ++t) {
      #pragma unroll
      for (int v = 0; v < 8; ++v)
        Yout[(((size_t)rt * 16 + mbase + v) << 9) + nb0 + t * 16] = y[t][v] * sc[v];
    }
  } else {
    // blank_prob = relu(enc + bdec) . w_blank + b_blank  -> out[:,0]
    float pb[8];
    #pragma unroll
    for (int v = 0; v < 8; ++v) {
      size_t row = (size_t)rt * 16 + mbase + v;
      float p = 0.0f;
      #pragma unroll
      for (int t = 0; t < 4; ++t) {
        int n = nb0 + t * 16;
        float s = Yother[(row << 9) + n] + y[t][v] * sc[v];
        p += fmaxf(s, 0.0f) * w_blank[n];
      }
      p += __shfl_xor(p, 1, 32);
      p += __shfl_xor(p, 2, 32);
      p += __shfl_xor(p, 4, 32);
      p += __shfl_xor(p, 8, 32);
      pb[v] = p;
    }
    if ((lane & 15) == 0) {
      #pragma unroll
      for (int v = 0; v < 8; ++v) red[mbase + v][wave] = pb[v];
    }
    __syncthreads();
    if (threadIdx.x < 16) {
      float s = 0.0f;
      #pragma unroll
      for (int w = 0; w < 8; ++w) s += red[threadIdx.x][w];
      out[((size_t)rt * 16 + threadIdx.x) * VFULL] = s + b_blank[0];
    }
  }
}

// ---- 512 -> 2048(pad) GEMM for the vocab heads ----
// grid (row_tiles, 4 col-slabs of 512). mode 0: raw logits -> region2
// mode 1: out[:,1+n] = gemm + b + (logit - lse) ; region2 = logit - lse
__global__ __launch_bounds__(256) void gemm_vocab(
    const unsigned short* __restrict__ As,
    const unsigned short* __restrict__ Wb,
    const float* __restrict__ lin_b,
    float* __restrict__ region2,
    const float* __restrict__ lse,
    float* __restrict__ out,
    int mode)
{
  int lane = threadIdx.x & 31;
  int wave = threadIdx.x >> 5;
  int rt = blockIdx.x;
  int slab = blockIdx.y;

  v8f acc[4] = {};
  const unsigned short* Ab = As + ((size_t)(rt * KT) << 9);
  int ntb = slab * 32 + wave * 4;
  #pragma unroll 2
  for (int kt = 0; kt < KT; ++kt) {
    v16bf a = *((const v16bf*)(Ab + ((size_t)kt << 9)) + lane);
    #pragma unroll
    for (int t = 0; t < 4; ++t) {
      v16bf b = *((const v16bf*)(Wb + ((size_t)((ntb + t) * KT + kt) << 9)) + lane);
      acc[t] = __builtin_amdgcn_wmma_f32_16x16x32_bf16(
          false, a, false, b, (short)0, acc[t], false, false);
    }
  }

  int mbase = (lane >> 4) << 3;
  int nb0 = slab * 512 + wave * 64 + (lane & 15);
  #pragma unroll
  for (int t = 0; t < 4; ++t) {
    int n = nb0 + t * 16;
    if (n < OV) {
      float lb = lin_b[n];
      #pragma unroll
      for (int v = 0; v < 8; ++v) {
        size_t row = (size_t)rt * 16 + mbase + v;
        float val = acc[t][v] + lb;
        size_t p2 = row * OV + n;
        if (mode == 0) {
          region2[p2] = val;
        } else {
          float lg = region2[p2] - lse[row];
          out[row * VFULL + 1 + n] = val + lg;
          region2[p2] = lg;
        }
      }
    }
  }
}

// ---- per-row logsumexp over 1999 logits ----
__global__ __launch_bounds__(256) void row_lse(const float* __restrict__ region2,
                                               float* __restrict__ lse) {
  __shared__ float red[256];
  int tid = threadIdx.x;
  const float* x = region2 + (size_t)blockIdx.x * OV;
  float mx = -3.402823466e38f;
  for (int i = tid; i < OV; i += 256) mx = fmaxf(mx, x[i]);
  red[tid] = mx; __syncthreads();
  for (int s = 128; s > 0; s >>= 1) {
    if (tid < s) red[tid] = fmaxf(red[tid], red[tid + s]);
    __syncthreads();
  }
  mx = red[0]; __syncthreads();
  float sm = 0.0f;
  for (int i = tid; i < OV; i += 256) sm += __expf(x[i] - mx);
  red[tid] = sm; __syncthreads();
  for (int s = 128; s > 0; s >>= 1) {
    if (tid < s) red[tid] += red[tid + s];
    __syncthreads();
  }
  if (tid == 0) lse[blockIdx.x] = mx + __logf(red[0]);
}

extern "C" void kernel_launch(void* const* d_in, const int* in_sizes, int n_in,
                              void* d_out, int out_size, void* d_ws, size_t ws_size,
                              hipStream_t stream) {
  const float* enc_in   = (const float*)d_in[0];
  const float* bdec_in  = (const float*)d_in[1];
  const float* vdec_in  = (const float*)d_in[2];
  const float* W_enc    = (const float*)d_in[3];
  const float* b_enc    = (const float*)d_in[4];
  const float* bn_enc_b = (const float*)d_in[5];
  const float* bn_enc_ls= (const float*)d_in[6];
  const float* W_bdec   = (const float*)d_in[7];
  const float* b_bdec   = (const float*)d_in[8];
  const float* bn_bdec_b = (const float*)d_in[9];
  const float* bn_bdec_ls= (const float*)d_in[10];
  const float* bn_vdec_b = (const float*)d_in[11];
  const float* bn_vdec_ls= (const float*)d_in[12];
  const float* W_blank  = (const float*)d_in[13];
  const float* b_blank  = (const float*)d_in[14];
  const float* W_ev     = (const float*)d_in[15];
  const float* b_ev     = (const float*)d_in[16];
  const float* W_dv     = (const float*)d_in[17];
  const float* b_dv     = (const float*)d_in[18];
  (void)in_sizes; (void)n_in; (void)out_size; (void)ws_size;

  float* out = (float*)d_out;                        // [32000, 2000]
  float* region2 = out + (size_t)R_TOTAL * VFULL;    // [32000, 1999] vocab_decoder_prob

  char* ws = (char*)d_ws;
  size_t off = 0;
  auto take = [&](size_t bytes) -> char* {
    char* p = ws + off;
    off += (bytes + 255) & ~(size_t)255;
    return p;
  };

  unsigned short* As_enc  = (unsigned short*)take((size_t)R_TOTAL * CDIM * 2);
  unsigned short* As_bdec = (unsigned short*)take((size_t)R_TOTAL * CDIM * 2); // later reused for swizzled enc
  unsigned short* As_vdec = (unsigned short*)take((size_t)R_TOTAL * CDIM * 2);
  float*          Yenc    = (float*)take((size_t)R_TOTAL * CDIM * 4);
  unsigned short* Wb_enc  = (unsigned short*)take((size_t)CDIM * CDIM * 2);
  unsigned short* Wb_bdec = (unsigned short*)take((size_t)CDIM * CDIM * 2);
  unsigned short* Wb_ev   = (unsigned short*)take((size_t)NPADV * CDIM * 2);
  unsigned short* Wb_dv   = (unsigned short*)take((size_t)NPADV * CDIM * 2);
  float*          lse     = (float*)take((size_t)R_TOTAL * 4);

  // weights -> bf16 B-swizzled
  swizzle_w<<<dim3((CDIM * CDIM) / 256), 256, 0, stream>>>(W_enc, Wb_enc, CDIM, CDIM);
  swizzle_w<<<dim3((CDIM * CDIM) / 256), 256, 0, stream>>>(W_bdec, Wb_bdec, CDIM, CDIM);
  swizzle_w<<<dim3((NPADV * CDIM) / 256), 256, 0, stream>>>(W_ev, Wb_ev, OV, NPADV);
  swizzle_w<<<dim3((NPADV * CDIM) / 256), 256, 0, stream>>>(W_dv, Wb_dv, OV, NPADV);

  // activations -> bf16 A-swizzled (vdec with fused pre-GEMM BiasNorm)
  dim3 gA(R_TOTAL / 8);
  swizzle_a<<<gA, 256, 0, stream>>>(enc_in,  As_enc,  nullptr, nullptr, 0);
  swizzle_a<<<gA, 256, 0, stream>>>(bdec_in, As_bdec, nullptr, nullptr, 0);
  swizzle_a<<<gA, 256, 0, stream>>>(vdec_in, As_vdec, bn_vdec_b, bn_vdec_ls, 1);

  // enc = BiasNorm(enc_in @ W_enc^T + b_enc)   (fp32 to ws)
  gemm512_bn<<<dim3(R_TOTAL / 16), 256, 0, stream>>>(
      As_enc, Wb_enc, b_enc, bn_enc_b, bn_enc_ls, Yenc,
      nullptr, nullptr, nullptr, nullptr, 0);

  // bdec GEMM+BiasNorm, fused blank head -> out[:,0]
  gemm512_bn<<<dim3(R_TOTAL / 16), 256, 0, stream>>>(
      As_bdec, Wb_bdec, b_bdec, bn_bdec_b, bn_bdec_ls, nullptr,
      Yenc, W_blank, b_blank, out, 1);

  // re-swizzle normalized enc (bf16) for W_ev GEMM; As_bdec region is free now
  swizzle_a<<<gA, 256, 0, stream>>>(Yenc, As_bdec, nullptr, nullptr, 0);

  // raw vocab-decoder logits -> region2
  gemm_vocab<<<dim3(R_TOTAL / 16, 4), 256, 0, stream>>>(
      As_vdec, Wb_dv, b_dv, region2, nullptr, nullptr, 0);

  // per-row logsumexp
  row_lse<<<dim3(R_TOTAL), 256, 0, stream>>>(region2, lse);

  // vocab_prob = enc@W_ev^T + b_ev + log_softmax(logits); also normalize region2 in place
  gemm_vocab<<<dim3(R_TOTAL / 16, 4), 256, 0, stream>>>(
      As_bdec, Wb_ev, b_ev, region2, lse, out, 1);
}
